// PredictorLG_22119081574782
// MI455X (gfx1250) — matrix operand order repaired and checked by reading, not compiled, not run on hardware
//
#include <hip/hip_runtime.h>
#include <hip/hip_bf16.h>
#include <math.h>

typedef __attribute__((ext_vector_type(16))) _Float16 v16h;
typedef __attribute__((ext_vector_type(8)))  _Float16 v8h;
typedef __attribute__((ext_vector_type(8)))  float    v8f;

#define EMBED 128
#define HALF  64

// Branchless exact-GELU: erf via Abramowitz-Stegun 7.1.26 (|err| < 1.5e-7),
// one v_rcp_f32 + one v_exp_f32, no EXEC divergence.
__device__ __forceinline__ float gelu_exact(float x) {
    float z  = fabsf(x) * 0.70710678118654752f;          // |x|/sqrt(2)
    float t  = __builtin_amdgcn_rcpf(fmaf(0.3275911f, z, 1.0f));
    float p  = fmaf(t, 1.061405429f, -1.453152027f);
    p = fmaf(t, p, 1.421413741f);
    p = fmaf(t, p, -0.284496736f);
    p = fmaf(t, p, 0.254829592f);
    p = p * t;
    float e  = __expf(-z * z);
    float er = fmaf(-p, e, 1.0f);                        // erf(z), z >= 0
    er = copysignf(er, x);
    return 0.5f * x * (1.0f + er);
}

// Branchless tanh: saturates correctly for large |x| (exp->inf/0 => +/-1).
__device__ __forceinline__ float tanh_fast(float x) {
    float e = __expf(2.0f * x);
    return 1.0f - 2.0f * __builtin_amdgcn_rcpf(e + 1.0f);
}

// ---------------------------------------------------------------------------
// Pack W_in (128x64) and W_out1 (128x64) into WMMA f16 B-fragment layout.
// B 32x16 fragment, wave32: lane holds col n = nt*16 + (lane&15),
// element i holds K = kc*32 + 8*(lane>>4) + (i < 8 ? i : i + 8).
// dst layout: [weight][tile = kc*4+nt][lane] of v16h (32 B each).
// ---------------------------------------------------------------------------
__global__ void pack_weights_kernel(const float* __restrict__ Win,
                                    const float* __restrict__ Wout1,
                                    _Float16* __restrict__ dst) {
    int idx = blockIdx.x * blockDim.x + threadIdx.x;  // 0..1023
    if (idx >= 1024) return;
    int w    = idx >> 9;
    int tile = (idx >> 5) & 15;
    int lane = idx & 31;
    int kc = tile >> 2, nt = tile & 3;
    const float* W = w ? Wout1 : Win;
    int n  = nt * 16 + (lane & 15);
    int kb = kc * 32 + 8 * (lane >> 4);
    v16h frag;
#pragma unroll
    for (int i = 0; i < 16; ++i) frag[i] = (_Float16)W[(kb + (i < 8 ? i : i + 8)) * HALF + n];
    ((v16h*)dst)[w * 512 + tile * 32 + lane] = frag;
}

// ---------------------------------------------------------------------------
// Guidance branch: per batch, LayerNorm -> 128x64 matmul -> GELU. Tiny (256
// tokens total), done in fp32 VALU. One block (128 threads) per batch.
// ---------------------------------------------------------------------------
__global__ void guidance_kernel(const float* __restrict__ gsrc,
                                const float* __restrict__ g,
                                const float* __restrict__ b,
                                const float* __restrict__ Wgd,
                                float* __restrict__ out) {
    __shared__ float rsum[EMBED];
    __shared__ float rsq[EMBED];
    __shared__ float normed[EMBED];
    int bidx = blockIdx.x, t = threadIdx.x;
    float v = gsrc[bidx * EMBED + t];
    rsum[t] = v; rsq[t] = v * v;
    __syncthreads();
    for (int o = EMBED / 2; o > 0; o >>= 1) {
        if (t < o) { rsum[t] += rsum[t + o]; rsq[t] += rsq[t + o]; }
        __syncthreads();
    }
    float mean = rsum[0] * (1.0f / EMBED);
    float var  = rsq[0] * (1.0f / EMBED) - mean * mean;
    float rstd = rsqrtf(var + 1e-5f);
    normed[t] = (v - mean) * rstd * g[t] + b[t];
    __syncthreads();
    if (t < HALF) {
        float acc = 0.f;
#pragma unroll 8
        for (int k = 0; k < EMBED; ++k) acc += normed[k] * Wgd[k * HALF + t];
        out[bidx * HALF + t] = gelu_exact(acc);
    }
}

// ---------------------------------------------------------------------------
// Main fused kernel: one wave (32 lanes) per 16-token tile.
//   LN(x) -> WMMA (128x64) -> GELU -> [concat guidance] -> WMMA (128x64)
//   -> GELU -> dot(64) -> tanh
// ---------------------------------------------------------------------------
__global__ void __launch_bounds__(256)
predictor_main_kernel(const float* __restrict__ x,
                      const float* __restrict__ ln_g,
                      const float* __restrict__ ln_b,
                      const _Float16* __restrict__ wpack,   // packed W_in | W_out1
                      const float* __restrict__ guid,       // [B][64] gelu'd guidance
                      const float* __restrict__ w2,         // [64] W_out2
                      float* __restrict__ out,
                      int numTiles, int tilesPerBatch) {
    __shared__ __attribute__((aligned(16))) _Float16 ldsbuf[8][16 * 72];

    const int lane = threadIdx.x & 31;
    const int wave = threadIdx.x >> 5;
    const int tile = blockIdx.x * 8 + wave;
    if (tile >= numTiles) return;                 // wave-uniform
    const int hl = lane >> 4;                     // half-wave select
    const int m  = lane & 15;                     // matrix row (A) / col (B)
    const int batch = tile / tilesPerBatch;
    const size_t tokBase = (size_t)tile * 16;

    // ---- Stage 1: LayerNorm fused into A-fragment build (registers only) ----
    // Lane l holds, per 32-K chunk c, K = 32c + {0..7, 16..23} + 8*hl of row m.
    // Lanes l and l^16 partition the row -> one shfl_xor(16) finishes stats.
    const float* xrow = x + (tokBase + m) * EMBED;
    float xs[4][16];
    float s = 0.f, sq = 0.f;
#pragma unroll
    for (int c = 0; c < 4; ++c) {
        const float4* pa = (const float4*)(xrow + 32 * c + 8 * hl);
        const float4* pb = (const float4*)(xrow + 32 * c + 16 + 8 * hl);
        float4 a0 = pa[0], a1 = pa[1], b0 = pb[0], b1 = pb[1];
        xs[c][0]=a0.x;  xs[c][1]=a0.y;  xs[c][2]=a0.z;  xs[c][3]=a0.w;
        xs[c][4]=a1.x;  xs[c][5]=a1.y;  xs[c][6]=a1.z;  xs[c][7]=a1.w;
        xs[c][8]=b0.x;  xs[c][9]=b0.y;  xs[c][10]=b0.z; xs[c][11]=b0.w;
        xs[c][12]=b1.x; xs[c][13]=b1.y; xs[c][14]=b1.z; xs[c][15]=b1.w;
#pragma unroll
        for (int i = 0; i < 16; ++i) { s += xs[c][i]; sq += xs[c][i] * xs[c][i]; }
    }
    s  += __shfl_xor(s, 16, 32);
    sq += __shfl_xor(sq, 16, 32);
    float mean = s * (1.0f / EMBED);
    float rstd = rsqrtf(sq * (1.0f / EMBED) - mean * mean + 1e-5f);

    v16h A[4];
#pragma unroll
    for (int c = 0; c < 4; ++c) {
        const float4* ga = (const float4*)(ln_g + 32 * c + 8 * hl);
        const float4* gb = (const float4*)(ln_g + 32 * c + 16 + 8 * hl);
        const float4* ba = (const float4*)(ln_b + 32 * c + 8 * hl);
        const float4* bb = (const float4*)(ln_b + 32 * c + 16 + 8 * hl);
        float4 g0 = ga[0], g1 = ga[1], g2 = gb[0], g3 = gb[1];
        float4 b0 = ba[0], b1 = ba[1], b2 = bb[0], b3 = bb[1];
        float gv[16] = {g0.x,g0.y,g0.z,g0.w, g1.x,g1.y,g1.z,g1.w,
                        g2.x,g2.y,g2.z,g2.w, g3.x,g3.y,g3.z,g3.w};
        float bv[16] = {b0.x,b0.y,b0.z,b0.w, b1.x,b1.y,b1.z,b1.w,
                        b2.x,b2.y,b2.z,b2.w, b3.x,b3.y,b3.z,b3.w};
#pragma unroll
        for (int i = 0; i < 16; ++i)
            A[c][i] = (_Float16)((xs[c][i] - mean) * rstd * gv[i] + bv[i]);
    }

    // ---- Stage 1 GEMM: local = gelu(LN(x) @ W_in), write f16 to LDS ----
    const v16h* Wt1 = (const v16h*)wpack;          // W_in tiles [0..511]
    _Float16* L = ldsbuf[wave];                    // 16 rows x 72 halves
#pragma unroll
    for (int nt = 0; nt < 4; ++nt) {
        v8f acc = {0.f,0.f,0.f,0.f,0.f,0.f,0.f,0.f};
#pragma unroll
        for (int c = 0; c < 4; ++c) {
            v16h bf = Wt1[(c * 4 + nt) * 32 + lane];
            acc = __builtin_amdgcn_wmma_f32_16x16x32_f16(
                false, A[c], false, bf, (short)0, acc, false, false);
        }
        // D layout: VGPR r -> row r + 8*hl, col nt*16 + m
#pragma unroll
        for (int r = 0; r < 8; ++r) {
            float v = gelu_exact(acc[r]);
            L[(r + 8 * hl) * 72 + nt * 16 + m] = (_Float16)v;
        }
    }
    __syncthreads();   // order cross-lane LDS store->load (block-uniform)

    // ---- Stage 2 A fragments: [local | guidance] ----
    v16h A2[4];
#pragma unroll
    for (int c = 0; c < 2; ++c) {                  // local halves from LDS
        int kb = 32 * c;
        v8h lo = *(const v8h*)&L[m * 72 + kb + 8 * hl];
        v8h hi = *(const v8h*)&L[m * 72 + kb + 16 + 8 * hl];
#pragma unroll
        for (int i = 0; i < 8; ++i) { A2[c][i] = lo[i]; A2[c][8 + i] = hi[i]; }
    }
    const float* G = guid + (size_t)batch * HALF;  // batch-uniform broadcast half
#pragma unroll
    for (int c = 0; c < 2; ++c) {
        const float4* pa = (const float4*)(G + 32 * c + 8 * hl);
        const float4* pb = (const float4*)(G + 32 * c + 16 + 8 * hl);
        float4 a0 = pa[0], a1 = pa[1], b0 = pb[0], b1 = pb[1];
        float gv[16] = {a0.x,a0.y,a0.z,a0.w, a1.x,a1.y,a1.z,a1.w,
                        b0.x,b0.y,b0.z,b0.w, b1.x,b1.y,b1.z,b1.w};
#pragma unroll
        for (int i = 0; i < 16; ++i) A2[2 + c][i] = (_Float16)gv[i];
    }

    // ---- Stage 2 GEMM + GELU + Stage 3 dot(W_out2) ----
    const v16h* Wt2 = (const v16h*)wpack + 512;    // W_out1 tiles
    float w2v[4];
#pragma unroll
    for (int nt = 0; nt < 4; ++nt) w2v[nt] = w2[nt * 16 + m];
    float part[8];
#pragma unroll
    for (int r = 0; r < 8; ++r) part[r] = 0.f;
#pragma unroll
    for (int nt = 0; nt < 4; ++nt) {
        v8f acc = {0.f,0.f,0.f,0.f,0.f,0.f,0.f,0.f};
#pragma unroll
        for (int c = 0; c < 4; ++c) {
            v16h bf = Wt2[(c * 4 + nt) * 32 + lane];
            acc = __builtin_amdgcn_wmma_f32_16x16x32_f16(
                false, A2[c], false, bf, (short)0, acc, false, false);
        }
#pragma unroll
        for (int r = 0; r < 8; ++r) part[r] += gelu_exact(acc[r]) * w2v[nt];
    }
    // Reduce over n within each 16-lane group (butterfly stays inside group).
#pragma unroll
    for (int r = 0; r < 8; ++r) {
        float p = part[r];
        p += __shfl_xor(p, 1, 32);
        p += __shfl_xor(p, 2, 32);
        p += __shfl_xor(p, 4, 32);
        p += __shfl_xor(p, 8, 32);
        part[r] = p;
    }
    float myv = 0.f;
#pragma unroll
    for (int r = 0; r < 8; ++r) myv = (m == r) ? part[r] : myv;
    if (m < 8) out[tokBase + m + 8 * hl] = tanh_fast(myv);
}

// ---------------------------------------------------------------------------
extern "C" void kernel_launch(void* const* d_in, const int* in_sizes, int n_in,
                              void* d_out, int out_size, void* d_ws, size_t ws_size,
                              hipStream_t stream) {
    const float* x       = (const float*)d_in[0];
    const float* gsrc    = (const float*)d_in[1];
    const float* ln_in_g = (const float*)d_in[2];
    const float* ln_in_b = (const float*)d_in[3];
    const float* W_in    = (const float*)d_in[4];
    const float* ln_gd_g = (const float*)d_in[5];
    const float* ln_gd_b = (const float*)d_in[6];
    const float* W_gd    = (const float*)d_in[7];
    const float* W_out1  = (const float*)d_in[8];
    const float* W_out2  = (const float*)d_in[9];

    const int B = in_sizes[1] / EMBED;                 // 256
    const int M = in_sizes[0] / EMBED;                 // total tokens (262144)
    const int N = M / B;                               // 1024

    _Float16* wpack = (_Float16*)d_ws;                 // 32 KB packed weights
    float* guidOut  = (float*)((char*)d_ws + 64 * 1024); // [B][64] fp32

    pack_weights_kernel<<<dim3(4), dim3(256), 0, stream>>>(W_in, W_out1, wpack);
    guidance_kernel<<<dim3(B), dim3(EMBED), 0, stream>>>(gsrc, ln_gd_g, ln_gd_b,
                                                         W_gd, guidOut);
    const int tiles = M / 16;
    const int blocks = (tiles + 7) / 8;
    predictor_main_kernel<<<dim3(blocks), dim3(256), 0, stream>>>(
        x, ln_in_g, ln_in_b, wpack, guidOut, W_out2, (float*)d_out,
        tiles, N / 16);
}
